// NodeGINe_24850680775301
// MI455X (gfx1250) — matrix operand order repaired and checked by reading, not compiled, not run on hardware
//
#include <hip/hip_runtime.h>

// ---------------- problem constants (match reference) ----------------
#define GN  100000
#define GE  1600000
#define GH  100
#define GL  3

typedef __attribute__((ext_vector_type(16))) _Float16 v16h;
typedef __attribute__((ext_vector_type(8)))  float    v8f;
typedef __attribute__((ext_vector_type(4)))  uint32_t u32x4;
typedef __attribute__((ext_vector_type(8)))  uint32_t u32x8;

__device__ __forceinline__ _Float16 f2h(float f) { return (_Float16)f; }

// low 32 bits of a generic pointer to LDS == LDS byte offset (ISA 10.2 aperture rules)
__device__ __forceinline__ uint32_t lds_lo(const void* p) {
    return (uint32_t)(uintptr_t)p;
}

// ---------------------------------------------------------------------
// TDM: 2-D tensor load Global->LDS (ISA ch.8 descriptor layout).
// tensor = [dim1 rows x dim0 cols] f32, row-major contiguous; tile = whole
// tensor. Issued by one wave; tracked on TENSORcnt.
// ---------------------------------------------------------------------
__device__ __forceinline__ void tdm_load_2d_f32(const void* gptr, uint32_t ldsoff,
                                                uint32_t dim0, uint32_t dim1)
{
    uint64_t ga = (uint64_t)gptr;
    u32x4 g0;
    g0[0] = 1u;                                          // count=1, user descriptor
    g0[1] = ldsoff;                                      // lds_addr
    g0[2] = (uint32_t)ga;                                // global_addr[31:0]
    g0[3] = (uint32_t)((ga >> 32) & 0x01FFFFFFu)         // global_addr[56:32]
          | (2u << 30);                                  // type=2 ("image")
    u32x8 g1;
    g1[0] = 0x00020000u;                                 // data_size=2 (4 bytes)
    g1[1] = (dim0 & 0xFFFFu) << 16;                      // tensor_dim0[15:0]
    g1[2] = (dim0 >> 16) | ((dim1 & 0xFFFFu) << 16);     // dim0[31:16] | dim1[15:0]
    g1[3] = (dim1 >> 16) | ((dim0 & 0xFFFFu) << 16);     // dim1[31:16] | tile_dim0
    g1[4] = (dim1 & 0xFFFFu);                            // tile_dim1 (tile_dim2=0)
    g1[5] = dim0;                                        // tensor_dim0_stride[31:0]
    g1[6] = 0u;                                          // stride hi | dim1_stride lo
    g1[7] = 0u;
    asm volatile("tensor_load_to_lds %0, %1" :: "s"(g0), "s"(g1) : "memory");
}

// async Global->LDS, per-lane 16B / 4B (tracked on ASYNCcnt)
__device__ __forceinline__ void async_g2l_b128(uint32_t ldsoff, uint32_t goff,
                                               const float* base)
{
    asm volatile("global_load_async_to_lds_b128 %0, %1, %2"
                 :: "v"(ldsoff), "v"(goff), "s"(base) : "memory");
}
__device__ __forceinline__ void async_g2l_b32(uint32_t ldsoff, uint32_t goff,
                                              const float* base)
{
    asm volatile("global_load_async_to_lds_b32 %0, %1, %2"
                 :: "v"(ldsoff), "v"(goff), "s"(base) : "memory");
}
__device__ __forceinline__ void wait_async0() {
    asm volatile("s_wait_asynccnt 0x0" ::: "memory");
}

// ---------------------------------------------------------------------
// Generic GEMM:  Out[M,Kout] = act( (A (+A2)) [M,Kin] * W[Kin,Kout] + bias )
//   W tile  : TDM tensor_load_to_lds   (TENSORcnt)
//   A tile  : global_load_async_to_lds (ASYNCcnt)
//   compute : v_wmma_f32_16x16x32_f16, A/B fragments in contiguous k = 16*(lane/16) + s order (wave32)
// Block: 128 threads (4 waves), 64 M-rows per block.
// ---------------------------------------------------------------------
template <int KINP, int KOUTP>
__launch_bounds__(128)
__global__ void gemm_wmma(const float* __restrict__ A,
                          const float* __restrict__ A2,
                          const float* __restrict__ W,
                          const float* __restrict__ bias,
                          float* __restrict__ Out,
                          int M, int Kin, int Kout, int relu)
{
    constexpr int NT = KOUTP / 16;
    __shared__ __align__(16) float    sW[KINP * KOUTP];   // raw W staging (TDM)
    __shared__ __align__(16) float    sA[64 * KINP];      // raw A staging (async)
    __shared__ __align__(16) _Float16 lA[64 * KINP];      // [row][k]  f16
    __shared__ __align__(16) _Float16 lB[KOUTP * KINP];   // [n][k]    f16 (W^T)

    const int tid  = threadIdx.x;
    const int row0 = blockIdx.x * 64;

    // ---- stage W via Tensor Data Mover (one wave issues; EXEC-independent) ----
    if (tid == 0) {
        tdm_load_2d_f32(W, lds_lo(sW), (uint32_t)Kout, (uint32_t)Kin);
        __builtin_amdgcn_s_wait_tensorcnt(0);
    }

    // ---- stage 64 rows of A via async global->LDS ----
    if (KINP == 128) {                       // rows are 16B aligned (Kin=128/100)
        const int cpr = Kin >> 2;            // float4 chunks per row
        for (int idx = tid; idx < 64 * cpr; idx += blockDim.x) {
            int r = idx / cpr, c = idx % cpr;
            int gr = row0 + r;
            if (gr < M) {
                async_g2l_b128(lds_lo(&sA[r * KINP + c * 4]),
                               (uint32_t)(((size_t)gr * Kin + (size_t)c * 4) * 4), A);
            } else {
                sA[r * KINP + c * 4 + 0] = 0.0f;
                sA[r * KINP + c * 4 + 1] = 0.0f;
                sA[r * KINP + c * 4 + 2] = 0.0f;
                sA[r * KINP + c * 4 + 3] = 0.0f;
            }
        }
    } else {                                 // small head GEMMs: 4B granules
        for (int idx = tid; idx < 64 * Kin; idx += blockDim.x) {
            int r = idx / Kin, k = idx % Kin;
            int gr = row0 + r;
            if (gr < M) {
                async_g2l_b32(lds_lo(&sA[r * KINP + k]),
                              (uint32_t)(((size_t)gr * Kin + k) * 4), A);
            } else {
                sA[r * KINP + k] = 0.0f;
            }
        }
    }
    wait_async0();
    __syncthreads();

    // ---- convert staged tiles to f16 (transpose W; fuse +A2 into A) ----
    for (int idx = tid; idx < KOUTP * KINP; idx += blockDim.x) {
        int n = idx / KINP, k = idx % KINP;
        float v = (n < Kout && k < Kin) ? sW[k * Kout + n] : 0.0f;
        lB[idx] = f2h(v);
    }
    for (int idx = tid; idx < 64 * KINP; idx += blockDim.x) {
        int r = idx / KINP, k = idx % KINP;
        int gr = row0 + r;
        float v = 0.0f;
        if (gr < M && k < Kin) {
            v = sA[r * KINP + k];
            if (A2) v += A2[(size_t)gr * Kin + k];
        }
        lA[idx] = f2h(v);
    }
    __syncthreads();

    // ---- WMMA main loop ----
    const int wave = tid >> 5;
    const int lane = tid & 31;
    const int mrow = lane & 15;
    const int hiK  = (lane >> 4) & 1;

    v8f acc[NT] = {};
    const uint32_t* aRow = (const uint32_t*)&lA[(wave * 16 + mrow) * KINP];

    for (int k0 = 0; k0 < KINP; k0 += 32) {
        union { v16h v; uint32_t u[8]; } af;
        const int ab = (k0 >> 1) + hiK * 8;
#pragma unroll
        for (int v = 0; v < 4; ++v) af.u[v]     = aRow[ab + v];
#pragma unroll
        for (int v = 0; v < 4; ++v) af.u[4 + v] = aRow[ab + 4 + v];
#pragma unroll
        for (int t = 0; t < NT; ++t) {
            const uint32_t* bRow = (const uint32_t*)&lB[(t * 16 + mrow) * KINP];
            const int bb = (k0 >> 1) + hiK * 8;
            union { v16h v; uint32_t u[8]; } bfr;
#pragma unroll
            for (int v = 0; v < 8; ++v) bfr.u[v] = bRow[bb + v];
            acc[t] = __builtin_amdgcn_wmma_f32_16x16x32_f16(
                false, af.v, false, bfr.v, (short)0, acc[t], false, false);
        }
    }

    // ---- epilogue: bias + optional relu, bounds-checked store ----
#pragma unroll
    for (int t = 0; t < NT; ++t) {
        int n = t * 16 + mrow;
        if (n >= Kout) continue;
        float b = bias ? bias[n] : 0.0f;
#pragma unroll
        for (int r = 0; r < 8; ++r) {
            int m = row0 + wave * 16 + hiK * 8 + r;
            if (m < M) {
                float v = acc[t][r] + b;
                if (relu && v < 0.0f) v = 0.0f;
                Out[(size_t)m * Kout + n] = v;
            }
        }
    }
}

// ---------------------------------------------------------------------
// Fused GINE edge kernel for one layer:
//   e        = ea[64-edge tile] @ lin_w + lin_b   (WMMA, in-register)
//   msg      = relu(xh[src] + e)
//   agg[dst] += msg                               (global_atomic_add_f32)
// Avoids materializing the 640 MB elin intermediate in HBM.
// ---------------------------------------------------------------------
__launch_bounds__(128)
__global__ void gine_edge_wmma(const float* __restrict__ ea,     // [E,H]
                               const float* __restrict__ lin_w,  // [H,H]
                               const float* __restrict__ lin_b,  // [H]
                               const float* __restrict__ xh,     // [N,H]
                               const int*   __restrict__ src,
                               const int*   __restrict__ dst,
                               float* __restrict__ agg,          // [N,H]
                               int E, int H)
{
    constexpr int KINP  = 128;
    constexpr int KOUTP = 112;
    constexpr int NT    = KOUTP / 16;
    __shared__ __align__(16) float    sW[KINP * KOUTP];
    __shared__ __align__(16) float    sA[64 * KINP];
    __shared__ __align__(16) _Float16 lA[64 * KINP];
    __shared__ __align__(16) _Float16 lB[KOUTP * KINP];

    const int tid  = threadIdx.x;
    const int row0 = blockIdx.x * 64;

    if (tid == 0) {
        tdm_load_2d_f32(lin_w, lds_lo(sW), (uint32_t)H, (uint32_t)H);
        __builtin_amdgcn_s_wait_tensorcnt(0);
    }

    {   // async-stage 64 edge rows of ea (H=100 floats -> 25 float4 chunks, 16B aligned)
        const int cpr = H >> 2;
        for (int idx = tid; idx < 64 * cpr; idx += blockDim.x) {
            int r = idx / cpr, c = idx % cpr;
            int ge = row0 + r;
            if (ge < E) {
                async_g2l_b128(lds_lo(&sA[r * KINP + c * 4]),
                               (uint32_t)(((size_t)ge * H + (size_t)c * 4) * 4), ea);
            } else {
                sA[r * KINP + c * 4 + 0] = 0.0f;
                sA[r * KINP + c * 4 + 1] = 0.0f;
                sA[r * KINP + c * 4 + 2] = 0.0f;
                sA[r * KINP + c * 4 + 3] = 0.0f;
            }
        }
    }
    wait_async0();
    __syncthreads();

    for (int idx = tid; idx < KOUTP * KINP; idx += blockDim.x) {
        int n = idx / KINP, k = idx % KINP;
        float v = (n < H && k < H) ? sW[k * H + n] : 0.0f;
        lB[idx] = f2h(v);
    }
    for (int idx = tid; idx < 64 * KINP; idx += blockDim.x) {
        int r = idx / KINP, k = idx % KINP;
        lA[idx] = f2h(k < H ? sA[r * KINP + k] : 0.0f);
    }
    __syncthreads();

    const int wave = tid >> 5;
    const int lane = tid & 31;
    const int mrow = lane & 15;
    const int hiK  = (lane >> 4) & 1;

    v8f acc[NT] = {};
    const uint32_t* aRow = (const uint32_t*)&lA[(wave * 16 + mrow) * KINP];

    for (int k0 = 0; k0 < KINP; k0 += 32) {
        union { v16h v; uint32_t u[8]; } af;
        const int ab = (k0 >> 1) + hiK * 8;
#pragma unroll
        for (int v = 0; v < 4; ++v) af.u[v]     = aRow[ab + v];
#pragma unroll
        for (int v = 0; v < 4; ++v) af.u[4 + v] = aRow[ab + 4 + v];
#pragma unroll
        for (int t = 0; t < NT; ++t) {
            const uint32_t* bRow = (const uint32_t*)&lB[(t * 16 + mrow) * KINP];
            const int bb = (k0 >> 1) + hiK * 8;
            union { v16h v; uint32_t u[8]; } bfr;
#pragma unroll
            for (int v = 0; v < 8; ++v) bfr.u[v] = bRow[bb + v];
            acc[t] = __builtin_amdgcn_wmma_f32_16x16x32_f16(
                false, af.v, false, bfr.v, (short)0, acc[t], false, false);
        }
    }

    // fused epilogue: gather x[src] + e, relu, scatter-add to agg[dst]
#pragma unroll
    for (int t = 0; t < NT; ++t) {
        int n = t * 16 + mrow;
        if (n >= H) continue;
        float b = lin_b[n];
#pragma unroll
        for (int r = 0; r < 8; ++r) {
            int e = row0 + wave * 16 + hiK * 8 + r;
            if (e < E) {
                int s = src[e], d = dst[e];
                __builtin_prefetch(&xh[(size_t)s * H], 0, 1);   // global_prefetch_b8
                float v = acc[t][r] + b + xh[(size_t)s * H + n];
                if (v > 0.0f)
                    atomicAdd(&agg[(size_t)d * H + n], v);
            }
        }
    }
}

// ---------------------------------------------------------------------
// utility kernels
// ---------------------------------------------------------------------
__global__ void zero_f32(float* p, long long n)
{
    long long i = (long long)blockIdx.x * blockDim.x + threadIdx.x;
    if (i < n) p[i] = 0.0f;
}

// per-column sum / sumsq over rows, LDS reduction (ds_add_f32) + global atomics
__global__ void bn_stats(const float* __restrict__ hbuf, float* __restrict__ stats,
                         int N, int H)
{
    __shared__ float s[224];              // [0..111]=sum, [112..223]=sumsq
    for (int i = threadIdx.x; i < 224; i += blockDim.x) s[i] = 0.0f;
    __syncthreads();
    int r = blockIdx.x * blockDim.x + threadIdx.x;
    if (r < N) {
        const float* row = hbuf + (size_t)r * H;
        for (int h = 0; h < H; ++h) {
            float v = row[h];
            atomicAdd(&s[h], v);
            atomicAdd(&s[112 + h], v * v);
        }
    }
    __syncthreads();
    for (int i = threadIdx.x; i < H; i += blockDim.x) {
        atomicAdd(&stats[i],       s[i]);
        atomicAdd(&stats[112 + i], s[112 + i]);
    }
}

// x = 0.5 * (x + relu( gamma*(h-mean)*rsqrt(var+eps) + beta ))
__global__ void bn_residual(float* __restrict__ x, const float* __restrict__ hbuf,
                            const float* __restrict__ stats,
                            const float* __restrict__ g, const float* __restrict__ beta,
                            int N, int H)
{
    long long i = (long long)blockIdx.x * blockDim.x + threadIdx.x;
    if (i >= (long long)N * H) return;
    int col = (int)(i % H);
    float invN = 1.0f / (float)N;
    float mean = stats[col] * invN;
    float var  = stats[112 + col] * invN - mean * mean;
    float hn = g[col] * (hbuf[i] - mean) * rsqrtf(var + 1e-5f) + beta[col];
    if (hn < 0.0f) hn = 0.0f;
    x[i] = (x[i] + hn) * 0.5f;
}

// ---------------------------------------------------------------------
// host-side orchestration
// ---------------------------------------------------------------------
extern "C" void kernel_launch(void* const* d_in, const int* in_sizes, int n_in,
                              void* d_out, int out_size, void* d_ws, size_t ws_size,
                              hipStream_t stream)
{
    (void)in_sizes; (void)n_in; (void)out_size; (void)ws_size;

    const float* x_in      = (const float*)d_in[0];
    const float* edge_attr = (const float*)d_in[1];
    const float* node_w    = (const float*)d_in[2];
    const float* node_b    = (const float*)d_in[3];
    const float* edge_w    = (const float*)d_in[4];
    const float* edge_b    = (const float*)d_in[5];
    const float* lin_w     = (const float*)d_in[6];
    const float* lin_b     = (const float*)d_in[7];
    const float* m1_w      = (const float*)d_in[8];
    const float* m1_b      = (const float*)d_in[9];
    const float* m2_w      = (const float*)d_in[10];
    const float* m2_b      = (const float*)d_in[11];
    const float* bn_g      = (const float*)d_in[12];
    const float* bn_beta   = (const float*)d_in[13];
    const float* h1_w      = (const float*)d_in[14];
    const float* h1_b      = (const float*)d_in[15];
    const float* h2_w      = (const float*)d_in[16];
    const float* h2_b      = (const float*)d_in[17];
    const float* h3_w      = (const float*)d_in[18];
    const float* h3_b      = (const float*)d_in[19];
    const int*   eidx      = (const int*)d_in[20];
    const int*   src       = eidx;
    const int*   dst       = eidx + GE;
    float*       out       = (float*)d_out;

    // workspace carve-out
    char* ws = (char*)d_ws;
    size_t off = 0;
    auto carve = [&](size_t nfloats) -> float* {
        float* p = (float*)(ws + off);
        off = (off + nfloats * sizeof(float) + 255) & ~(size_t)255;
        return p;
    };
    float* xh    = carve((size_t)GN * GH);   // node features (updated per layer)
    float* ea    = carve((size_t)GE * GH);   // encoded edge attrs (reused each layer)
    float* agg   = carve((size_t)GN * GH);   // scatter accumulator
    float* t1    = carve((size_t)GN * GH);   // MLP hidden
    float* h2b   = carve((size_t)GN * GH);   // MLP out
    float* stats = carve(256);               // BN column sums / sumsq
    float* p1    = carve((size_t)GN * 50);
    float* p2    = carve((size_t)GN * 25);

    const dim3 blk(128);
    const int  gridN = (GN + 63) / 64;
    const int  gridE = (GE + 63) / 64;
    const long long NH = (long long)GN * GH;

    // encoders
    gemm_wmma<128, 112><<<gridN, blk, 0, stream>>>(x_in, nullptr, node_w, node_b,
                                                   xh, GN, 128, GH, 0);
    gemm_wmma<128, 112><<<gridE, blk, 0, stream>>>(edge_attr, nullptr, edge_w, edge_b,
                                                   ea, GE, 128, GH, 0);

    for (int i = 0; i < GL; ++i) {
        // zero scatter accumulator
        zero_f32<<<(int)((NH + 255) / 256), 256, 0, stream>>>(agg, NH);
        // fused: edge linear (WMMA) + gather + relu + scatter-add
        gine_edge_wmma<<<gridE, blk, 0, stream>>>(ea, lin_w + (size_t)i * GH * GH,
                                                  lin_b + (size_t)i * GH,
                                                  xh, src, dst, agg, GE, GH);
        // conv MLP: relu((x+agg)@m1+b1)@m2+b2   (A2 fused add inside GEMM)
        gemm_wmma<128, 112><<<gridN, blk, 0, stream>>>(xh, agg,
            m1_w + (size_t)i * GH * GH, m1_b + (size_t)i * GH, t1, GN, GH, GH, 1);
        gemm_wmma<128, 112><<<gridN, blk, 0, stream>>>(t1, nullptr,
            m2_w + (size_t)i * GH * GH, m2_b + (size_t)i * GH, h2b, GN, GH, GH, 0);
        // batchnorm (training-mode) + relu + averaged residual
        zero_f32<<<1, 256, 0, stream>>>(stats, 224);
        bn_stats<<<(GN + 255) / 256, 256, 0, stream>>>(h2b, stats, GN, GH);
        bn_residual<<<(int)((NH + 255) / 256), 256, 0, stream>>>(
            xh, h2b, stats, bn_g + (size_t)i * GH, bn_beta + (size_t)i * GH, GN, GH);
    }

    // classification head: 100 -> 50 -> 25 -> 2
    gemm_wmma<128, 64><<<gridN, blk, 0, stream>>>(xh, nullptr, h1_w, h1_b, p1, GN, 100, 50, 1);
    gemm_wmma<64, 32><<<gridN, blk, 0, stream>>>(p1, nullptr, h2_w, h2_b, p2, GN, 50, 25, 1);
    gemm_wmma<32, 16><<<gridN, blk, 0, stream>>>(p2, nullptr, h3_w, h3_b, out, GN, 25, 2, 0);
}
